// GraphStructuredAttention_76450417869073
// MI455X (gfx1250) — matrix operand order repaired and checked by reading, not compiled, not run on hardware
//
#include <hip/hip_runtime.h>
#include <hip/hip_bf16.h>

// ---------------------------------------------------------------------------
// Fused multi-head attention for MI455X (gfx1250), bf16 WMMA pipeline.
//   x:(4,2048,1024) f32, mask:(4,2048,2048) i32, W*:(1024,1024) f32, b*: f32
// Stages: convert/transpose -> QKV GEMM (WMMA, 16x64/wave, double-buffered,
// V stored transposed per head) -> flash attention (WMMA, software-pipelined
// K/V fragment loads) -> output GEMM (WMMA).  fp32 accumulation throughout.
// Workspace use: ~88 MB.
// ---------------------------------------------------------------------------

typedef __bf16 bf16;
typedef __bf16 v16bf __attribute__((ext_vector_type(16)));
typedef __bf16 v8bf  __attribute__((ext_vector_type(8)));
typedef float  v8f   __attribute__((ext_vector_type(8)));

#define D_MODEL 1024
#define N_HEADS 16
#define D_HEAD  64
#define BATCH   4
#define SEQ     2048
#define M_TOTAL (BATCH * SEQ)   // 8192 rows

union V16 { v16bf v; v8bf h[2]; bf16 e[16]; };
struct GFrag { V16 a; v16bf b[4]; };   // one k-step of a 16x64 GEMM tile

__device__ __forceinline__ float rowmax16(float v) {
#pragma unroll
  for (int i = 1; i < 16; i <<= 1) v = fmaxf(v, __shfl_xor(v, i, 32));
  return v;
}
__device__ __forceinline__ float rowsum16(float v) {
#pragma unroll
  for (int i = 1; i < 16; i <<= 1) v += __shfl_xor(v, i, 32);
  return v;
}

// ---------------- stage 0: fp32 -> bf16 converts -----------------------------
__global__ void cvt_bf16_kernel(const float* __restrict__ in,
                                bf16* __restrict__ out, int n) {
  int i = blockIdx.x * blockDim.x + threadIdx.x;
  int stride = gridDim.x * blockDim.x;
  for (; i < n; i += stride) out[i] = (bf16)in[i];
}

// W (K=1024 rows x N=1024 cols) -> Wt[n][k] bf16 (B-fragments contiguous in K)
__global__ void transpose_bf16_kernel(const float* __restrict__ W,
                                      bf16* __restrict__ Wt) {
  int i = blockIdx.x * blockDim.x + threadIdx.x;   // 1M elements
  int k = i >> 10, n = i & 1023;
  Wt[(size_t)n * D_MODEL + k] = (bf16)W[(size_t)k * D_MODEL + n];
}

// ---------------- stage 1: QKV projection GEMM (bf16 WMMA) -------------------
// One wave computes a 16x64 tile of Y = X @ W; fragment loads for step k+1 are
// issued before the WMMAs of step k (double-buffered software pipeline).
// Q,K written head-major (b,h,s,d); V written transposed per head (b,h,d,s).
__global__ void __launch_bounds__(256)
qkv_gemm_kernel(const bf16* __restrict__ xb,
                const bf16* __restrict__ WtQ, const bf16* __restrict__ WtK,
                const bf16* __restrict__ WtV,
                const float* __restrict__ bq, const float* __restrict__ bk,
                const float* __restrict__ bv,
                bf16* __restrict__ Qh, bf16* __restrict__ Kh,
                bf16* __restrict__ Vt) {
  const int wave = threadIdx.x >> 5, lane = threadIdx.x & 31;
  const int m0 = blockIdx.x * 16;
  const int n0 = (blockIdx.y * 8 + wave) * 64;   // 16x64 tile per wave

  const bf16* Wt; const float* bias; bf16* out; int vmode;
  if (blockIdx.z == 0)      { Wt = WtQ; bias = bq; out = Qh; vmode = 0; }
  else if (blockIdx.z == 1) { Wt = WtK; bias = bk; out = Kh; vmode = 0; }
  else                      { Wt = WtV; bias = bv; out = Vt; vmode = 1; }

  const int arow  = m0 + (lane & 15);
  const int akoff = (lane < 16) ? 0 : 16;   // A: K halves {0..15}/{16..31}
  const int bkoff = (lane < 16) ? 0 : 16;   // B: lane half selects K 0-15 / 16-31

  v8f acc[4] = {};

  auto load_frag = [&](int k0, GFrag& f) {
    const bf16* ap = xb + (size_t)arow * D_MODEL + k0 + akoff;
    f.a.h[0] = *(const v8bf*)ap;
    f.a.h[1] = *(const v8bf*)(ap + 8);
#pragma unroll
    for (int t = 0; t < 4; ++t)
      f.b[t] = *(const v16bf*)(Wt + (size_t)(n0 + t * 16 + (lane & 15)) *
                                        D_MODEL + k0 + bkoff);
  };
  auto compute = [&](const GFrag& f) {
#pragma unroll
    for (int t = 0; t < 4; ++t)
      acc[t] = __builtin_amdgcn_wmma_f32_16x16x32_bf16(
          false, f.a.v, false, f.b[t], (short)0, acc[t], false, false);
  };

  GFrag cur, nxt;
  load_frag(0, cur);
#pragma unroll 2
  for (int k0 = 0; k0 < D_MODEL - 32; k0 += 32) {
    load_frag(k0 + 32, nxt);   // in flight while current WMMAs execute
    compute(cur);
    cur = nxt;
  }
  compute(cur);

#pragma unroll
  for (int t = 0; t < 4; ++t) {
    const int ncol = n0 + t * 16 + (lane & 15);
    const int h = ncol >> 6, d = ncol & 63;
    const float bb = bias[ncol];
#pragma unroll
    for (int r = 0; r < 8; ++r) {
      int m = m0 + r + ((lane >> 4) << 3);  // C layout: row = r + 8*lanehalf
      int bi = m >> 11;                     // / 2048
      int s  = m & 2047;
      bf16 val = (bf16)(acc[t][r] + bb);
      if (vmode)
        out[(((size_t)bi * N_HEADS + h) * D_HEAD + d) * SEQ + s] = val;
      else
        out[(((size_t)bi * N_HEADS + h) * SEQ + s) * D_HEAD + d] = val;
    }
  }
}

// ---------------- stage 2: flash attention (bf16 WMMA) -----------------------
// One wave owns a 16-query tile of one (b,h); streams 32 keys / iteration.
// K fragments are prefetched one full iteration ahead; V fragments and mask
// words are loaded at the top of the iteration, overlapping the softmax VALU.
__global__ void __launch_bounds__(256)
attn_kernel(const bf16* __restrict__ Qh, const bf16* __restrict__ Kh,
            const bf16* __restrict__ Vt, const int* __restrict__ mask,
            bf16* __restrict__ attnb) {
  __attribute__((aligned(32))) __shared__ bf16 Plds[8][16 * 32];  // per-wave P

  const int wave = threadIdx.x >> 5, lane = threadIdx.x & 31;
  const int bh = blockIdx.y;                 // 0..63
  const int b  = bh >> 4;
  const int hh = bh & 15;
  const int q0 = (blockIdx.x * 8 + wave) * 16;

  const bf16* Qb  = Qh + (size_t)bh * SEQ * D_HEAD;
  const bf16* Kb  = Kh + (size_t)bh * SEQ * D_HEAD;
  const bf16* Vtb = Vt + (size_t)bh * D_HEAD * SEQ;   // [d][s]
  const int* mb   = mask + (size_t)b * SEQ * SEQ;
  const float scale = 0.125f;                // 1/sqrt(64)

  const int qrow_lo = (lane >> 4) << 3;      // C layout row offset (0 or 8)
  const int kcol    = lane & 15;

  // Q A-fragments for d-chunks [0,32) and [32,64)
  V16 qf[2];
#pragma unroll
  for (int c = 0; c < 2; ++c) {
    const bf16* qp = Qb + (size_t)(q0 + (lane & 15)) * D_HEAD + 32 * c +
                     ((lane < 16) ? 0 : 16);
    qf[c].h[0] = *(const v8bf*)qp;
    qf[c].h[1] = *(const v8bf*)(qp + 8);
  }

  v8f acc0 = {}, acc1 = {}, acc2 = {}, acc3 = {};
  float mrun[8], lrun[8];
#pragma unroll
  for (int r = 0; r < 8; ++r) { mrun[r] = -1e30f; lrun[r] = 0.f; }

  // K fragments [j][d-chunk], prefetched one iteration ahead
  v16bf kf[2][2];
  auto load_kf = [&](int kbase, v16bf (&k)[2][2]) {
#pragma unroll
    for (int j = 0; j < 2; ++j)
#pragma unroll
      for (int cc = 0; cc < 2; ++cc)
        k[j][cc] = *(const v16bf*)(Kb +
                                   (size_t)(kbase + j * 16 + kcol) * D_HEAD +
                                   32 * cc + ((lane < 16) ? 0 : 16));
  };
  load_kf(0, kf);

  for (int kt = 0; kt < SEQ / 32; ++kt) {
    const int kbase = kt * 32;

    // ---- early independent loads: V fragments + mask words ----
    v16bf vf[4];
    const int krow0 = kbase + ((lane < 16) ? 0 : 16);
#pragma unroll
    for (int t = 0; t < 4; ++t)
      vf[t] = *(const v16bf*)(Vtb + (size_t)(t * 16 + kcol) * SEQ + krow0);
    int mvv[2][8];
#pragma unroll
    for (int j = 0; j < 2; ++j) {
      const int k = kbase + j * 16 + kcol;
#pragma unroll
      for (int r = 0; r < 8; ++r)
        mvv[j][r] = mb[(size_t)(q0 + r + qrow_lo) * SEQ + k];
    }

    // ---- scores: two 16x16 tiles (keys j*16..j*16+15) ----
    v8f s[2];
#pragma unroll
    for (int j = 0; j < 2; ++j) {
      v8f c = {};
      c = __builtin_amdgcn_wmma_f32_16x16x32_bf16(false, qf[0].v, false,
                                                  kf[j][0], (short)0, c,
                                                  false, false);
      c = __builtin_amdgcn_wmma_f32_16x16x32_bf16(false, qf[1].v, false,
                                                  kf[j][1], (short)0, c,
                                                  false, false);
      s[j] = c;
    }

    // ---- prefetch next iteration's K fragments (one full iter of distance)
    if (kt + 1 < SEQ / 32) load_kf(kbase + 32, kf);

    // ---- mask + scale ----
#pragma unroll
    for (int j = 0; j < 2; ++j)
#pragma unroll
      for (int r = 0; r < 8; ++r) {
        float v = s[j][r] * scale;
        s[j][r] = mvv[j][r] ? v : -1e30f;
      }

    // ---- online softmax (row stats live per-lane for this lane's row) ----
    float scl[8];
#pragma unroll
    for (int r = 0; r < 8; ++r) {
      float rm   = rowmax16(fmaxf(s[0][r], s[1][r]));
      float mnew = fmaxf(mrun[r], rm);
      float f    = __expf(mrun[r] - mnew);
      float p0   = __expf(s[0][r] - mnew);
      float p1   = __expf(s[1][r] - mnew);
      s[0][r] = p0; s[1][r] = p1;
      lrun[r] = lrun[r] * f + rowsum16(p0 + p1);
      mrun[r] = mnew;
      scl[r]  = f;
    }
#pragma unroll
    for (int r = 0; r < 8; ++r) {
      acc0[r] *= scl[r]; acc1[r] *= scl[r];
      acc2[r] *= scl[r]; acc3[r] *= scl[r];
    }

    // ---- P: C layout -> LDS -> A layout (16x32 bf16) ----
#pragma unroll
    for (int j = 0; j < 2; ++j)
#pragma unroll
      for (int r = 0; r < 8; ++r)
        Plds[wave][(r + qrow_lo) * 32 + j * 16 + kcol] = (bf16)s[j][r];
    asm volatile("s_wait_dscnt 0" ::: "memory");   // wave-local LDS RAW fence
    V16 pf;
    {
      const bf16* pp = &Plds[wave][(lane & 15) * 32 + ((lane < 16) ? 0 : 16)];
      pf.h[0] = *(const v8bf*)pp;
      pf.h[1] = *(const v8bf*)(pp + 8);
    }

    // ---- O += P @ V : four 16x16 output tiles (d chunks of 16) ----
    acc0 = __builtin_amdgcn_wmma_f32_16x16x32_bf16(false, pf.v, false, vf[0],
                                                   (short)0, acc0, false, false);
    acc1 = __builtin_amdgcn_wmma_f32_16x16x32_bf16(false, pf.v, false, vf[1],
                                                   (short)0, acc1, false, false);
    acc2 = __builtin_amdgcn_wmma_f32_16x16x32_bf16(false, pf.v, false, vf[2],
                                                   (short)0, acc2, false, false);
    acc3 = __builtin_amdgcn_wmma_f32_16x16x32_bf16(false, pf.v, false, vf[3],
                                                   (short)0, acc3, false, false);
  }

  // ---- normalize and store heads-merged bf16 (B, S, D_MODEL) ----
#pragma unroll
  for (int r = 0; r < 8; ++r) {
    const float inv = 1.0f / lrun[r];
    const int q = q0 + r + qrow_lo;
    const size_t row = ((size_t)b * SEQ + q) * D_MODEL + hh * 64;
    attnb[row +  0 + kcol] = (bf16)(acc0[r] * inv);
    attnb[row + 16 + kcol] = (bf16)(acc1[r] * inv);
    attnb[row + 32 + kcol] = (bf16)(acc2[r] * inv);
    attnb[row + 48 + kcol] = (bf16)(acc3[r] * inv);
  }
}

// ---------------- stage 3: output GEMM out = attn @ Wo + bo (fp32 out) -------
__global__ void __launch_bounds__(256)
out_gemm_kernel(const bf16* __restrict__ attnb, const bf16* __restrict__ WtO,
                const float* __restrict__ bo, float* __restrict__ out) {
  const int wave = threadIdx.x >> 5, lane = threadIdx.x & 31;
  const int m0 = blockIdx.x * 16;
  const int n0 = (blockIdx.y * 8 + wave) * 64;   // 16x64 tile per wave
  const int arow  = m0 + (lane & 15);
  const int akoff = (lane < 16) ? 0 : 16;
  const int bkoff = (lane < 16) ? 0 : 16;

  v8f acc[4] = {};

  auto load_frag = [&](int k0, GFrag& f) {
    const bf16* ap = attnb + (size_t)arow * D_MODEL + k0 + akoff;
    f.a.h[0] = *(const v8bf*)ap;
    f.a.h[1] = *(const v8bf*)(ap + 8);
#pragma unroll
    for (int t = 0; t < 4; ++t)
      f.b[t] = *(const v16bf*)(WtO + (size_t)(n0 + t * 16 + (lane & 15)) *
                                         D_MODEL + k0 + bkoff);
  };
  auto compute = [&](const GFrag& f) {
#pragma unroll
    for (int t = 0; t < 4; ++t)
      acc[t] = __builtin_amdgcn_wmma_f32_16x16x32_bf16(
          false, f.a.v, false, f.b[t], (short)0, acc[t], false, false);
  };

  GFrag cur, nxt;
  load_frag(0, cur);
#pragma unroll 2
  for (int k0 = 0; k0 < D_MODEL - 32; k0 += 32) {
    load_frag(k0 + 32, nxt);
    compute(cur);
    cur = nxt;
  }
  compute(cur);

#pragma unroll
  for (int t = 0; t < 4; ++t) {
    const int ncol = n0 + t * 16 + (lane & 15);
    const float bias = bo[ncol];
#pragma unroll
    for (int r = 0; r < 8; ++r) {
      int m = m0 + r + ((lane >> 4) << 3);
      out[(size_t)m * D_MODEL + ncol] = acc[t][r] + bias;
    }
  }
}

// ---------------------------------------------------------------------------
extern "C" void kernel_launch(void* const* d_in, const int* in_sizes, int n_in,
                              void* d_out, int out_size, void* d_ws,
                              size_t ws_size, hipStream_t stream) {
  const float* x  = (const float*)d_in[0];
  const int*   mk = (const int*)d_in[1];
  const float* Wq = (const float*)d_in[2];
  const float* bq = (const float*)d_in[3];
  const float* Wk = (const float*)d_in[4];
  const float* bk = (const float*)d_in[5];
  const float* Wv = (const float*)d_in[6];
  const float* bv = (const float*)d_in[7];
  const float* Wo = (const float*)d_in[8];
  const float* bo = (const float*)d_in[9];
  float* out = (float*)d_out;

  // workspace partition (~88 MB)
  char* p = (char*)d_ws;
  const size_t XB = (size_t)M_TOTAL * D_MODEL * sizeof(bf16);   // 16 MB
  const size_t WB = (size_t)D_MODEL * D_MODEL * sizeof(bf16);   // 2 MB
  bf16* xb  = (bf16*)p; p += XB;
  bf16* WtQ = (bf16*)p; p += WB;
  bf16* WtK = (bf16*)p; p += WB;
  bf16* WtV = (bf16*)p; p += WB;
  bf16* WtO = (bf16*)p; p += WB;
  bf16* Qh  = (bf16*)p; p += XB;
  bf16* Kh  = (bf16*)p; p += XB;
  bf16* Vt  = (bf16*)p; p += XB;   // V transposed per head: (b,h,d,s)
  bf16* attnb = (bf16*)p; p += XB;

  // stage 0: converts
  cvt_bf16_kernel<<<8192, 256, 0, stream>>>(x, xb, M_TOTAL * D_MODEL);
  transpose_bf16_kernel<<<4096, 256, 0, stream>>>(Wq, WtQ);
  transpose_bf16_kernel<<<4096, 256, 0, stream>>>(Wk, WtK);
  transpose_bf16_kernel<<<4096, 256, 0, stream>>>(Wv, WtV);
  transpose_bf16_kernel<<<4096, 256, 0, stream>>>(Wo, WtO);

  // stage 1: Q/K/V projections (grid.z selects which; 16x64 tile per wave)
  qkv_gemm_kernel<<<dim3(M_TOTAL / 16, D_MODEL / 512, 3), 256, 0, stream>>>(
      xb, WtQ, WtK, WtV, bq, bk, bv, Qh, Kh, Vt);

  // stage 2: flash attention over 64 (b,h) pairs, 128 q-tiles each
  attn_kernel<<<dim3(16, BATCH * N_HEADS), 256, 0, stream>>>(Qh, Kh, Vt, mk,
                                                             attnb);

  // stage 3: output projection
  out_gemm_kernel<<<dim3(M_TOTAL / 16, D_MODEL / 512), 256, 0, stream>>>(
      attnb, WtO, bo, out);
}
